// Attention_57767310131800
// MI455X (gfx1250) — matrix operand order repaired and checked by reading, not compiled, not run on hardware
//
#include <hip/hip_runtime.h>
#include <hip/hip_bf16.h>

// ---------------- problem constants ----------------
#define BB 2
#define TT 2
#define HH 128
#define WW 128
#define CC 256
#define MIDC 256
#define NHEAD 8
#define HD 32
#define PATCH 8
#define STEP 6
#define NWH 21          // windows per spatial dim
#define NWIN (NWH*NWH)  // 441
#define LL 128          // T * 8 * 8

typedef __bf16 bf16x16 __attribute__((ext_vector_type(16)));
typedef __bf16 bf16x8  __attribute__((ext_vector_type(8)));
typedef float  f32x8   __attribute__((ext_vector_type(8)));
typedef int    i32x4   __attribute__((vector_size(16)));    // GCC-style, matches builtin

// ---------------- optional CDNA5 async global->LDS path ----------------
#if defined(__has_builtin)
#if __has_builtin(__builtin_amdgcn_global_load_async_to_lds_b128) && \
    __has_builtin(__builtin_amdgcn_s_wait_asynccnt)
#define USE_ASYNC 1
#endif
#endif

#ifdef USE_ASYNC
typedef __attribute__((address_space(1))) i32x4* gbl_v4i_ptr;
typedef __attribute__((address_space(3))) i32x4* lds_v4i_ptr;
static __device__ __forceinline__ void async_cp16(__bf16* lds_dst, const __bf16* gsrc) {
    // integer round-trips: strip const, do the addrspace conversion explicitly
    // (AS1 = flat 64-bit address; AS3 = low 32 bits of the generic LDS address)
    gbl_v4i_ptr g = (gbl_v4i_ptr)(unsigned long long)(uintptr_t)gsrc;
    lds_v4i_ptr l = (lds_v4i_ptr)(unsigned int)(uintptr_t)lds_dst;
    __builtin_amdgcn_global_load_async_to_lds_b128(g, l, 0, 0);
}
static __device__ __forceinline__ void async_wait0() {
    __builtin_amdgcn_s_wait_asynccnt(0);
}
#else
static __device__ __forceinline__ void async_cp16(__bf16* lds_dst, const __bf16* gsrc) {
    *(bf16x8*)lds_dst = *(const bf16x8*)gsrc;
}
static __device__ __forceinline__ void async_wait0() {}
#endif

// ---------------- WMMA helpers (CDNA5 gfx1250) ----------------
static __device__ __forceinline__ f32x8 bwmma(bf16x16 a, bf16x16 b, f32x8 c) {
    // D = A(16x32) * B(32x16) + C, f32 accumulate
    return __builtin_amdgcn_wmma_f32_16x16x32_bf16(false, a, false, b, (short)0, c, false, false);
}

// A fragment (16x32, 16-bit): lane m = lane&15; elems 0..7 -> K = hi*8..+7,
// elems 8..15 -> K = 16+hi*8..+7  (ISA 7.12.2 16-bit A layout)
static __device__ __forceinline__ bf16x16 load_a_frag(const __bf16* row, int hi) {
    bf16x8 lo = *(const bf16x8*)(row + hi * 8);
    bf16x8 h8 = *(const bf16x8*)(row + 16 + hi * 8);
    return __builtin_shufflevector(lo, h8, 0,1,2,3,4,5,6,7,8,9,10,11,12,13,14,15);
}

// B fragment (32x16, 16-bit): lane n = lane&15; elems 0..7 -> K = hi*8..+7, elems 8..15 -> K = 16+hi*8..+7
// "row" points at the length-32 K-vector belonging to column n of B
static __device__ __forceinline__ bf16x16 load_b_frag(const __bf16* row, int hi) {
    bf16x8 lo = *(const bf16x8*)(row + hi * 8);
    bf16x8 h8 = *(const bf16x8*)(row + 16 + hi * 8);
    return __builtin_shufflevector(lo, h8, 0,1,2,3,4,5,6,7,8,9,10,11,12,13,14,15);
}

static __device__ __forceinline__ float wave_sum32(float v) {
    #pragma unroll
    for (int m = 16; m >= 1; m >>= 1) v += __shfl_xor(v, m, 32);
    return v;
}
static __device__ __forceinline__ float half_sum16(float v) {
    #pragma unroll
    for (int m = 8; m >= 1; m >>= 1) v += __shfl_xor(v, m, 32);
    return v;
}
static __device__ __forceinline__ float half_max16(float v) {
    #pragma unroll
    for (int m = 8; m >= 1; m >>= 1) v = fmaxf(v, __shfl_xor(v, m, 32));
    return v;
}

// ---------------- Kernel 1: LayerNorm -> bf16 ----------------
__global__ __launch_bounds__(256) void ln_kernel(const float* __restrict__ x,
                                                 const float* __restrict__ g,
                                                 const float* __restrict__ b,
                                                 __bf16* __restrict__ xn) {
    const int wave = threadIdx.x >> 5;
    const int lane = threadIdx.x & 31;
    const size_t token = (size_t)blockIdx.x * 8 + wave;   // 65536 tokens
    const float* xp = x + token * CC + lane * 8;
    float v[8];
    *(float4*)(v)     = *(const float4*)(xp);
    *(float4*)(v + 4) = *(const float4*)(xp + 4);
    float s = 0.f;
    #pragma unroll
    for (int i = 0; i < 8; ++i) s += v[i];
    s = wave_sum32(s);
    const float mu = s * (1.0f / 256.0f);
    float vs = 0.f;
    #pragma unroll
    for (int i = 0; i < 8; ++i) { float d = v[i] - mu; vs += d * d; }
    vs = wave_sum32(vs);
    const float rs = rsqrtf(vs * (1.0f / 256.0f) + 1e-6f);
    float gv[8], bv[8];
    *(float4*)(gv)     = *(const float4*)(g + lane * 8);
    *(float4*)(gv + 4) = *(const float4*)(g + lane * 8 + 4);
    *(float4*)(bv)     = *(const float4*)(b + lane * 8);
    *(float4*)(bv + 4) = *(const float4*)(b + lane * 8 + 4);
    bf16x8 o;
    #pragma unroll
    for (int i = 0; i < 8; ++i) o[i] = (__bf16)((v[i] - mu) * rs * gv[i] + bv[i]);
    *(bf16x8*)(xn + token * CC + lane * 8) = o;
}

// ---------------- Kernel 2: f32 -> bf16 weight convert ----------------
__global__ __launch_bounds__(256) void wconv_kernel(const float* __restrict__ wq,
                                                    const float* __restrict__ wk,
                                                    const float* __restrict__ wv,
                                                    const float* __restrict__ wo,
                                                    __bf16* __restrict__ oq,
                                                    __bf16* __restrict__ ok,
                                                    __bf16* __restrict__ ov,
                                                    __bf16* __restrict__ oo) {
    const int idx = blockIdx.x * 256 + threadIdx.x;       // 262144 total
    const int which = idx >> 16;
    const int off = idx & 65535;
    const float* s = (which == 0) ? wq : (which == 1) ? wk : (which == 2) ? wv : wo;
    __bf16* d = (which == 0) ? oq : (which == 1) ? ok : (which == 2) ? ov : oo;
    d[off] = (__bf16)s[off];
}

// stage one head's Wq/Wk/Wv slices ([32][256] each) into LDS buffer
static __device__ __forceinline__ void stage_w(__bf16* dst,
                                               const __bf16* sq,
                                               const __bf16* sk,
                                               const __bf16* sv,
                                               int tid) {
    __bf16* dq = dst;
    __bf16* dk = dst + HD * CC;
    __bf16* dv = dst + 2 * HD * CC;
    #pragma unroll
    for (int e = tid; e < (HD * CC) / 8; e += 256) {      // 4 iters/thread
        async_cp16(dq + e * 8, sq + e * 8);
        async_cp16(dk + e * 8, sk + e * 8);
        async_cp16(dv + e * 8, sv + e * 8);
    }
}

// ---------------- Kernel 3: fused per-window attention (all 8 heads) ----------------
// grid 882, 256 threads. Dynamic LDS = 221184 B.
__global__ __launch_bounds__(256) void attn_kernel(const __bf16* __restrict__ xn,
                                                   const __bf16* __restrict__ wq,
                                                   const __bf16* __restrict__ wk,
                                                   const __bf16* __restrict__ wv,
                                                   __bf16* __restrict__ wout) {
    extern __shared__ char smem[];
    __bf16* Xs   = (__bf16*)smem;             // [128][256]                 32768
    __bf16* Wbuf = Xs + 128 * 256;            // 2 x (3 x [32][256])        49152
    __bf16* Qs   = Wbuf + 2 * 3 * HD * CC;    // [128][32] (also O bounce)   4096
    __bf16* Ks   = Qs + 128 * HD;             // [128][32]                   4096
    __bf16* VTs  = Ks + 128 * HD;             // [32][128] (V transposed)    4096
    __bf16* Ps   = VTs + HD * LL;             // [128][128]                 16384

    const int flat = blockIdx.x;              // reference flat = n_i*B + b_i
    const int b_i = flat & 1;
    const int n_i = flat >> 1;
    const int row0 = (n_i / NWH) * STEP;
    const int col0 = (n_i % NWH) * STEP;
    const int tid = threadIdx.x;
    const int wid = tid >> 5;
    const int lane = tid & 31;
    const int lm = lane & 15;
    const int hi = lane >> 4;

    // --- prologue: stage head-0 weights + X window (async if available) ---
    stage_w(Wbuf, wq, wk, wv, tid);
    {
        const int l = tid >> 1;
        const int c0 = (tid & 1) * 128;
        const int t = l >> 6, rem = l & 63, i = rem >> 3, j = rem & 7;
        const __bf16* src = xn + ((((size_t)((b_i * TT + t) * HH + row0 + i)) * WW + (col0 + j)) * CC) + c0;
        __bf16* dst = Xs + l * CC + c0;
        #pragma unroll
        for (int c = 0; c < 128; c += 8) async_cp16(dst + c, src + c);
    }
    async_wait0();
    __syncthreads();

    for (int head = 0; head < NHEAD; ++head) {
        const int buf = head & 1;
        // prefetch next head's weights into the other buffer (ASYNCcnt)
        if (head + 1 < NHEAD)
            stage_w(Wbuf + (buf ^ 1) * 3 * HD * CC,
                    wq + (size_t)(head + 1) * HD * CC,
                    wk + (size_t)(head + 1) * HD * CC,
                    wv + (size_t)(head + 1) * HD * CC, tid);

        const __bf16* Wqs = Wbuf + buf * 3 * HD * CC;
        const __bf16* Wks = Wqs + HD * CC;
        const __bf16* Wvs = Wks + HD * CC;

        // --- Q,K,V = X * W^T  ([128x256]x[256x32]); wave wid owns 16-row tile ---
        f32x8 qa[2] = {}, ka[2] = {}, va[2] = {};
        #pragma unroll
        for (int kk = 0; kk < 8; ++kk) {
            const int kb = kk * 32;
            const bf16x16 A = load_a_frag(Xs + (wid * 16 + lm) * CC + kb, hi);
            #pragma unroll
            for (int nt = 0; nt < 2; ++nt) {
                const bf16x16 Bq = load_b_frag(Wqs + (nt * 16 + lm) * CC + kb, hi);
                const bf16x16 Bk = load_b_frag(Wks + (nt * 16 + lm) * CC + kb, hi);
                const bf16x16 Bv = load_b_frag(Wvs + (nt * 16 + lm) * CC + kb, hi);
                qa[nt] = bwmma(A, Bq, qa[nt]);
                ka[nt] = bwmma(A, Bk, ka[nt]);
                va[nt] = bwmma(A, Bv, va[nt]);
            }
        }
        #pragma unroll
        for (int nt = 0; nt < 2; ++nt) {
            #pragma unroll
            for (int vg = 0; vg < 8; ++vg) {
                const int m = wid * 16 + vg + hi * 8;   // token row
                const int n = nt * 16 + lm;             // hd column
                Qs[m * HD + n]  = (__bf16)qa[nt][vg];
                Ks[m * HD + n]  = (__bf16)ka[nt][vg];
                VTs[n * LL + m] = (__bf16)va[nt][vg];   // transposed
            }
        }
        __syncthreads();   // b1: Q/K/VT ready

        // --- S = Q * K^T kept in registers (8 N-tiles, K=hd=32 single step) ---
        f32x8 s[8];
        #pragma unroll
        for (int nt = 0; nt < 8; ++nt) s[nt] = (f32x8){};
        {
            const bf16x16 Aq = load_a_frag(Qs + (wid * 16 + lm) * HD, hi);
            #pragma unroll
            for (int nt = 0; nt < 8; ++nt) {
                // B[k=c][n=m'] = K[m'][c] -> contiguous row read of Ks
                const bf16x16 Bk = load_b_frag(Ks + (nt * 16 + lm) * HD, hi);
                s[nt] = bwmma(Aq, Bk, s[nt]);
            }
        }
        // --- softmax over 128 columns (8 frags x 16 lanes per row) ---
        const float scale = 0.17677669529663687f;  // 1/sqrt(32)
        #pragma unroll
        for (int vg = 0; vg < 8; ++vg) {
            float mx = -3.0e38f;
            #pragma unroll
            for (int nt = 0; nt < 8; ++nt) { s[nt][vg] *= scale; mx = fmaxf(mx, s[nt][vg]); }
            mx = half_max16(mx);
            float sum = 0.f;
            #pragma unroll
            for (int nt = 0; nt < 8; ++nt) { float e = __expf(s[nt][vg] - mx); s[nt][vg] = e; sum += e; }
            sum = half_sum16(sum);
            const float inv = 1.0f / sum;
            const int r = wid * 16 + vg + hi * 8;
            #pragma unroll
            for (int nt = 0; nt < 8; ++nt)
                Ps[r * LL + nt * 16 + lm] = (__bf16)(s[nt][vg] * inv);
        }
        __syncthreads();   // b2: P ready; Qs/Ks free from here

        // --- O = P * V  ([128x128]x[128x32], K loop = 4) ---
        f32x8 o[2] = {};
        #pragma unroll
        for (int kk = 0; kk < 4; ++kk) {
            const int kb = kk * 32;
            const bf16x16 Ap = load_a_frag(Ps + (wid * 16 + lm) * LL + kb, hi);
            #pragma unroll
            for (int nt = 0; nt < 2; ++nt) {
                // B[k=m'][n=c] = V[m'][c] = VT[c][m'] -> contiguous row read of VTs
                const bf16x16 Bv = load_b_frag(VTs + (nt * 16 + lm) * LL + kb, hi);
                o[nt] = bwmma(Ap, Bv, o[nt]);
            }
        }
        // bounce O tile through Qs region for coalesced global stores
        #pragma unroll
        for (int nt = 0; nt < 2; ++nt) {
            #pragma unroll
            for (int vg = 0; vg < 8; ++vg) {
                const int m = wid * 16 + vg + hi * 8;
                Qs[m * HD + nt * 16 + lm] = (__bf16)o[nt][vg];
            }
        }
        __syncthreads();   // b3: O tile complete
        {
            const int r = tid >> 1;
            const int c = (tid & 1) * 16;
            const bf16x8 v0 = *(const bf16x8*)(Qs + r * HD + c);
            const bf16x8 v1 = *(const bf16x8*)(Qs + r * HD + c + 8);
            __bf16* d = wout + ((size_t)flat * LL + r) * MIDC + head * HD + c;
            *(bf16x8*)d = v0;
            *(bf16x8*)(d + 8) = v1;
        }
        if (head + 1 < NHEAD) async_wait0();   // next W buffer landed
        __syncthreads();   // b4: safe to overwrite Qs/Ks/VTs/Ps, W buffer visible
    }
}

// ---------------- Kernel 4: overlap-add gather + out projection ----------------
// grid 512, 256 threads, dynamic LDS = 196608 B. Each block: 128 pixels.
__global__ __launch_bounds__(256) void outproj_kernel(const __bf16* __restrict__ wout,
                                                      const __bf16* __restrict__ wo,
                                                      const float* __restrict__ bias,
                                                      float* __restrict__ out) {
    extern __shared__ char smem[];
    __bf16* Xs  = (__bf16*)smem;         // [128][256] rec tile (bf16)
    __bf16* Wos = Xs + 128 * 256;        // [256][256] Wo (reused as f32 [128][256])

    const int tid = threadIdx.x;
    for (int e = tid; e < (CC * MIDC) / 8; e += 256)
        async_cp16(Wos + e * 8, wo + e * 8);

    // --- gather + count-normalize into LDS ---
    {
        const int pr = tid >> 1;
        const int c0 = (tid & 1) * 128;
        const int p = blockIdx.x * 128 + pr;             // flat (b,t,h,w)
        const int w = p & 127;
        const int h = (p >> 7) & 127;
        const int t = (p >> 14) & 1;
        const int bo_ = p >> 15;
        int ihs[2], iws[2], nih = 0, niw = 0;
        {
            int i1 = h / STEP; if (i1 > NWH - 1) i1 = NWH - 1;
            int i0 = (h >= 7) ? (h - 2) / STEP : 0;      // ceil((h-7)/6)
            for (int i = i0; i <= i1; ++i) if (h - STEP * i < PATCH) ihs[nih++] = i;
            i1 = w / STEP; if (i1 > NWH - 1) i1 = NWH - 1;
            i0 = (w >= 7) ? (w - 2) / STEP : 0;
            for (int i = i0; i <= i1; ++i) if (w - STEP * i < PATCH) iws[niw++] = i;
        }
        const float inv = 1.0f / (float)(nih * niw);
        for (int c = 0; c < 128; c += 8) {
            float acc[8] = {0.f, 0.f, 0.f, 0.f, 0.f, 0.f, 0.f, 0.f};
            for (int a = 0; a < nih; ++a) {
                for (int b2 = 0; b2 < niw; ++b2) {
                    const int nwid = ihs[a] * NWH + iws[b2];
                    const int l = t * 64 + (h - STEP * ihs[a]) * 8 + (w - STEP * iws[b2]);
                    const bf16x8 v = *(const bf16x8*)(wout +
                        ((size_t)(bo_ * NWIN + nwid) * LL + l) * MIDC + c0 + c);
                    #pragma unroll
                    for (int k = 0; k < 8; ++k) acc[k] += (float)v[k];
                }
            }
            bf16x8 ov;
            #pragma unroll
            for (int k = 0; k < 8; ++k) ov[k] = (__bf16)(acc[k] * inv);
            *(bf16x8*)(Xs + pr * CC + c0 + c) = ov;
        }
    }
    async_wait0();
    __syncthreads();

    // --- out = rec * Wo^T + bo  ([128x256]x[256x256]) ---
    const int wid = tid >> 5;
    const int lane = tid & 31;
    const int lm = lane & 15;
    const int hi = lane >> 4;
    f32x8 acc[16];
    #pragma unroll
    for (int nt = 0; nt < 16; ++nt) acc[nt] = (f32x8){};
    #pragma unroll
    for (int kk = 0; kk < 8; ++kk) {
        const int kb = kk * 32;
        const bf16x16 A = load_a_frag(Xs + (wid * 16 + lm) * CC + kb, hi);
        #pragma unroll
        for (int nt = 0; nt < 16; ++nt) {
            // B[k=m][n=c] = Wo[c][m] -> contiguous row read of Wos (Wo is [C][MID])
            const bf16x16 Bf = load_b_frag(Wos + (nt * 16 + lm) * MIDC + kb, hi);
            acc[nt] = bwmma(A, Bf, acc[nt]);
        }
    }
    __syncthreads();                         // all waves done reading Xs/Wos
    float* Fs = (float*)Wos;                 // reuse as f32 [128][256] (128 KB)
    #pragma unroll
    for (int nt = 0; nt < 16; ++nt) {
        const float bv = bias[nt * 16 + lm];
        #pragma unroll
        for (int vg = 0; vg < 8; ++vg) {
            const int m = wid * 16 + vg + hi * 8;
            Fs[m * CC + nt * 16 + lm] = acc[nt][vg] + bv;
        }
    }
    __syncthreads();                         // f32 tile complete
    {
        const int r = tid >> 1;
        const int c0 = (tid & 1) * 128;
        const float* src = Fs + r * CC + c0;
        float* d = out + ((size_t)blockIdx.x * 128 + r) * CC + c0;
        #pragma unroll
        for (int i = 0; i < 128; i += 4)
            *(float4*)(d + i) = *(const float4*)(src + i);
    }
}

// ---------------- launcher ----------------
extern "C" void kernel_launch(void* const* d_in, const int* in_sizes, int n_in,
                              void* d_out, int out_size, void* d_ws, size_t ws_size,
                              hipStream_t stream) {
    const float* x   = (const float*)d_in[0];
    const float* lng = (const float*)d_in[1];
    const float* lnb = (const float*)d_in[2];
    const float* Wq  = (const float*)d_in[3];
    const float* Wk  = (const float*)d_in[4];
    const float* Wv  = (const float*)d_in[5];
    const float* Wo  = (const float*)d_in[6];
    const float* bo  = (const float*)d_in[7];

    char* ws = (char*)d_ws;
    __bf16* xn  = (__bf16*)ws;                                  // 33,554,432 B
    __bf16* wqb = (__bf16*)(ws + 33554432);                     // 131,072 B each
    __bf16* wkb = (__bf16*)(ws + 33554432 + 131072);
    __bf16* wvb = (__bf16*)(ws + 33554432 + 2 * 131072);
    __bf16* wob = (__bf16*)(ws + 33554432 + 3 * 131072);
    __bf16* wout = (__bf16*)(ws + 33554432 + 4 * 131072);       // 57,802,752 B

    ln_kernel<<<8192, 256, 0, stream>>>(x, lng, lnb, xn);
    wconv_kernel<<<1024, 256, 0, stream>>>(Wq, Wk, Wv, Wo, wqb, wkb, wvb, wob);

    const size_t attn_lds = (size_t)(128 * 256 + 2 * 3 * 32 * 256 + 2 * 128 * 32 + 32 * 128 + 128 * 128) * 2; // 221,184
    attn_kernel<<<882, 256, attn_lds, stream>>>(xn, wqb, wkb, wvb, wout);

    const size_t proj_lds = (size_t)(128 * 256 + 256 * 256) * 2; // 196,608
    outproj_kernel<<<512, 256, proj_lds, stream>>>(wout, wob, bo, (float*)d_out);
}
